// Block_68771016343957
// MI455X (gfx1250) — matrix operand order repaired and checked by reading, not compiled, not run on hardware
//
#include <hip/hip_runtime.h>
#include <cstdint>
#include <cstddef>

// ---- problem constants (match reference) ----
#define B_   2
#define T_   2048
#define D_   1024
#define H_   16
#define HD_  64
#define E_   8
#define BT_  (B_*T_)

// ---- WMMA fragment types (gfx1250, wave32) ----
typedef __attribute__((ext_vector_type(16))) __bf16          v16bf;
typedef __attribute__((ext_vector_type(16))) unsigned short  v16u;
typedef __attribute__((ext_vector_type(8)))  float           v8f;

union frag_u { uint4 q[2]; v16u v; unsigned short e[16]; };

static __device__ __forceinline__ unsigned short f2bf(float f) {
  unsigned int u = __float_as_uint(f);
  u += 0x7FFFu + ((u >> 16) & 1u);          // round-to-nearest-even
  return (unsigned short)(u >> 16);
}

// ---- gfx1250 async global->LDS staging (probe-verified signature) ----
#if defined(__gfx1250__) && __has_builtin(__builtin_amdgcn_global_load_async_to_lds_b128)
#define USE_ASYNC_LDS 1
typedef int v4i_vs __attribute__((vector_size(16)));
static __device__ __forceinline__ void async_cp16(const unsigned short* g, unsigned short* l) {
  __builtin_amdgcn_global_load_async_to_lds_b128((v4i_vs*)g, (v4i_vs*)l, 0, 0);
}
static __device__ __forceinline__ void async_wait0() {
#if __has_builtin(__builtin_amdgcn_s_wait_asynccnt)
  __builtin_amdgcn_s_wait_asynccnt(0);
#else
  asm volatile("s_wait_asynccnt 0" ::: "memory");
#endif
}
#endif

// =====================================================================
// fp32 -> bf16 transposing convert: dst[n*K + k] = bf16(src[k*N + n])
// =====================================================================
__global__ __launch_bounds__(256) void k_cvt_t(const float* __restrict__ s,
                                               unsigned short* __restrict__ d,
                                               int K, int N) {
  s += (size_t)blockIdx.z * K * N;
  d += (size_t)blockIdx.z * K * N;
  __shared__ float tile[32][33];
  const int kt = blockIdx.x * 32, nt = blockIdx.y * 32;
  const int tx = threadIdx.x & 31, ty = threadIdx.x >> 5;
  #pragma unroll
  for (int i = ty; i < 32; i += 8)
    tile[i][tx] = s[(size_t)(kt + i) * N + nt + tx];
  __syncthreads();
  #pragma unroll
  for (int i = ty; i < 32; i += 8)
    d[(size_t)(nt + i) * K + kt + tx] = f2bf(tile[tx][i]);
}

// =====================================================================
// LayerNorm: one block per token row; bf16 out (+ optional fp32 copy)
// =====================================================================
__global__ __launch_bounds__(256) void k_ln(const float* __restrict__ x,
                                            const float* __restrict__ w,
                                            unsigned short* __restrict__ obf,
                                            float* __restrict__ of) {
  const int row = blockIdx.x;
  const float* xr = x + (size_t)row * D_;
  __shared__ float red[256];
  const int tid = threadIdx.x;

  float s = 0.f;
  for (int d = tid; d < D_; d += 256) s += xr[d];
  red[tid] = s; __syncthreads();
  for (int o = 128; o > 0; o >>= 1) { if (tid < o) red[tid] += red[tid + o]; __syncthreads(); }
  const float mu = red[0] * (1.0f / D_);
  __syncthreads();

  float v = 0.f;
  for (int d = tid; d < D_; d += 256) { float c = xr[d] - mu; v += c * c; }
  red[tid] = v; __syncthreads();
  for (int o = 128; o > 0; o >>= 1) { if (tid < o) red[tid] += red[tid + o]; __syncthreads(); }
  const float rstd = rsqrtf(red[0] * (1.0f / D_) + 1e-5f);

  for (int d = tid; d < D_; d += 256) {
    const float o = w[d] * (xr[d] - mu) * rstd;
    obf[(size_t)row * D_ + d] = f2bf(o);
    if (of) of[(size_t)row * D_ + d] = o;
  }
}

// =====================================================================
// bf16 WMMA GEMM, double-buffered async pipeline.
// C[M,N] = act(A[M,K] @ B[K,N] (+res)); B given TRANSPOSED as Bt[N,K].
// Block tile 64x128x32; 4 waves, each 32x64 (2x4 WMMA tiles, 8 wmma/iter).
// LDS rows padded to 40 elems (20-bank stride, conflict-free b128).
// Pipeline: issue async loads for tile k+1, do WMMAs on tile k, then
// s_wait_asynccnt + barrier -> DMA overlaps the math.
// =====================================================================
#define BM   64
#define BN   128
#define BKK  32
#define LSTR 40

__global__ __launch_bounds__(128) void k_gemm_bf16(
    const unsigned short* __restrict__ A, int lda,   // [M,K]
    const unsigned short* __restrict__ Bt, int ldb,  // [N,K]  (ldb == K)
    float* __restrict__ Cf, unsigned short* __restrict__ Cb, int ldc,
    int M, int N, int K,
    const int* __restrict__ gather,
    const int* __restrict__ scatter,
    const float* __restrict__ scale,
    const int* __restrict__ cntp,
    const float* __restrict__ residual,
    int do_gelu)
{
  const int Meff = cntp ? *cntp : M;
  const int m0 = blockIdx.x * BM;
  const int n0 = blockIdx.y * BN;
  if (m0 >= Meff) return;

  __shared__ __align__(32) unsigned short As[2 * BM * LSTR];  // [buf][m][k]
  __shared__ __align__(32) unsigned short Bs[2 * BN * LSTR];  // [buf][n][k]

  const int tid  = threadIdx.x;
  const int lane = tid & 31;
  const int wave = tid >> 5;
  const int wm = (wave >> 1) * 32;
  const int wn = (wave & 1) * 64;
  const int hi  = (lane >> 4) & 1;
  const int l16 = lane & 15;

  v8f acc[2][4] = {};

  // A stager: thread -> (row 0..63, 16-elem k-chunk). B stager: row = tid.
  const int sr = tid >> 1;
  const int sc = (tid & 1) * 16;
  const int gm_l = m0 + sr;
  const bool avalid = (gm_l < Meff);
  long long arow = 0;
  if (avalid) arow = gather ? gather[gm_l] : gm_l;

  // zero-fill invalid A rows once (both buffers stay zero)
  if (!avalid) {
    uint4 z = {0u,0u,0u,0u};
    #pragma unroll
    for (int buf = 0; buf < 2; ++buf) {
      uint4* dst = (uint4*)(As + buf * (BM * LSTR) + sr * LSTR + sc);
      dst[0] = z; dst[1] = z;
    }
  }

  auto stage = [&](int buf, int k0) {
    unsigned short* Ab = As + buf * (BM * LSTR);
    unsigned short* Bb = Bs + buf * (BN * LSTR);
    const unsigned short* ag = A  + (size_t)arow * lda + k0 + sc;
    const unsigned short* bg = Bt + (size_t)(n0 + tid) * ldb + k0;
#ifdef USE_ASYNC_LDS
    if (avalid) {
      async_cp16(ag,     Ab + sr * LSTR + sc);
      async_cp16(ag + 8, Ab + sr * LSTR + sc + 8);
    }
    #pragma unroll
    for (int c = 0; c < 4; ++c)
      async_cp16(bg + c * 8, Bb + tid * LSTR + c * 8);
#else
    if (avalid) {
      ((uint4*)(Ab + sr * LSTR + sc))[0] = ((const uint4*)ag)[0];
      ((uint4*)(Ab + sr * LSTR + sc))[1] = ((const uint4*)ag)[1];
    }
    #pragma unroll
    for (int c = 0; c < 4; ++c)
      ((uint4*)(Bb + tid * LSTR + c * 8))[0] = ((const uint4*)(bg + c * 8))[0];
#endif
    if (k0 + BKK < K) {   // global_prefetch_b8 for the tile after next
      if (avalid) __builtin_prefetch(ag + BKK, 0, 0);
      __builtin_prefetch(bg + BKK, 0, 0);
    }
  };

  const int nk = K / BKK;
  stage(0, 0);
#ifdef USE_ASYNC_LDS
  async_wait0();
#endif
  __syncthreads();

  for (int kb = 0; kb < nk; ++kb) {
    const int cur = kb & 1;
    if (kb + 1 < nk) stage(cur ^ 1, (kb + 1) * BKK);   // overlap with WMMAs

    const unsigned short* Ab = As + cur * (BM * LSTR);
    const unsigned short* Bb = Bs + cur * (BN * LSTR);

    v16bf af[2], bfr[4];
    #pragma unroll
    for (int i = 0; i < 2; ++i) {
      const unsigned short* p = Ab + (wm + i * 16 + l16) * LSTR + hi * 8;
      frag_u u;
      u.q[0] = ((const uint4*)p)[0];
      u.q[1] = ((const uint4*)(p + 16))[0];
      af[i] = __builtin_bit_cast(v16bf, u.v);
    }
    #pragma unroll
    for (int j = 0; j < 4; ++j) {
      const unsigned short* p = Bb + (wn + j * 16 + l16) * LSTR + hi * 8;
      frag_u u;
      u.q[0] = ((const uint4*)p)[0];
      u.q[1] = ((const uint4*)(p + 16))[0];
      bfr[j] = __builtin_bit_cast(v16bf, u.v);
    }
    #pragma unroll
    for (int i = 0; i < 2; ++i)
      #pragma unroll
      for (int j = 0; j < 4; ++j)
        acc[i][j] = __builtin_amdgcn_wmma_f32_16x16x32_bf16(
            false, af[i], false, bfr[j], (short)0, acc[i][j], false, false);

#ifdef USE_ASYNC_LDS
    async_wait0();      // next tile's DMA has overlapped the math above
#endif
    __syncthreads();
  }

  // epilogue
  #pragma unroll
  for (int i = 0; i < 2; ++i) {
    #pragma unroll
    for (int j = 0; j < 4; ++j) {
      #pragma unroll
      for (int r = 0; r < 8; ++r) {
        const int gm = m0 + wm + i * 16 + r + hi * 8;
        const int gn = n0 + wn + j * 16 + l16;
        if (gm < Meff) {
          float v = acc[i][j][r];
          if (residual) v += residual[(size_t)gm * ldc + gn];
          if (do_gelu)  v = 0.5f * v * (1.0f + erff(v * 0.70710678118f));
          if (scatter)       Cf[(size_t)scatter[gm] * ldc + gn] += scale[gm] * v;
          else if (Cb)       Cb[(size_t)gm * ldc + gn] = f2bf(v);
          else               Cf[(size_t)gm * ldc + gn] = v;
        }
      }
    }
  }
}

// =====================================================================
// Flash attention: 4 waves/block = 4 consecutive 16-query tiles (uniform
// causal bound). K staged via async DMA (stride-72 rows), V staged
// TRANSPOSED (Vt[d][key], stride 40); all fragments load as b128 pairs.
// =====================================================================
#define KSTR 72
#define VSTR 40

__global__ __launch_bounds__(128) void k_attn(
    const unsigned short* __restrict__ qkv,  // [B*T, 3D] bf16
    const float* __restrict__ gate,          // [B*T]
    unsigned short* __restrict__ y)          // [B*T, D] bf16
{
  const int QB = T_ / 64;
  const int q64 = blockIdx.x % QB;
  const int h   = (blockIdx.x / QB) % H_;
  const int b   = blockIdx.x / (QB * H_);
  const int tid  = threadIdx.x;
  const int lane = tid & 31;
  const int wave = tid >> 5;
  const int hi  = lane >> 4;
  const int l16 = lane & 15;
  const int qbase = q64 * 64 + wave * 16;
  const size_t RS = 3 * D_;
  const unsigned short* qp = qkv + (size_t)b * T_ * RS + h * HD_;
  const unsigned short* kp = qp + D_;
  const unsigned short* vp = qp + 2 * D_;

  __shared__ __align__(32) unsigned short Ks[32 * KSTR]; // [key][d]
  __shared__ __align__(32) unsigned short Vt[64 * VSTR]; // [d][key]
  __shared__ float Pt[4][16][33];

  // Q fragments: rows qbase..+15, two 32-wide d-chunks, d = 8*hi+s (s<8), 16+8*hi+(s-8) (s>=8)
  v16bf aq[2];
  {
    const unsigned short* qrow = qp + (size_t)(qbase + l16) * RS;
    #pragma unroll
    for (int c = 0; c < 2; ++c) {
      frag_u u;
      u.q[0] = ((const uint4*)(qrow + c * 32 + hi * 8))[0];
      u.q[1] = ((const uint4*)(qrow + c * 32 + 16 + hi * 8))[0];
      aq[c] = __builtin_bit_cast(v16bf, u.v);
    }
  }

  float Mr[8], Lr[8], lg[8];
  v8f O[4] = {};
  #pragma unroll
  for (int r = 0; r < 8; ++r) {
    Mr[r] = -3.0e38f; Lr[r] = 0.0f;
    const float g = gate[(size_t)b * T_ + qbase + r + hi * 8];
    lg[r] = logf(fminf(fmaxf(g, 1e-6f), 1.0f));
  }

  const int nkb = q64 * 2 + 2;
  for (int kb = 0; kb < nkb; ++kb) {
    const int k0 = kb * 32;
    // stage K tile [32][64] via async DMA
    {
      const int kr = tid >> 2, kc = (tid & 3) * 16;
      const unsigned short* src = kp + (size_t)(k0 + kr) * RS + kc;
      unsigned short* dst = Ks + kr * KSTR + kc;
#ifdef USE_ASYNC_LDS
      async_cp16(src, dst); async_cp16(src + 8, dst + 8);
#else
      ((uint4*)dst)[0] = ((const uint4*)src)[0];
      ((uint4*)dst)[1] = ((const uint4*)src)[1];
#endif
    }
    // stage V tile transposed -> Vt[d][key]
    {
      const int vr = tid >> 2, vc = (tid & 3) * 16;
      frag_u tv;
      const uint4* src = (const uint4*)(vp + (size_t)(k0 + vr) * RS + vc);
      tv.q[0] = src[0]; tv.q[1] = src[1];
      #pragma unroll
      for (int i = 0; i < 16; ++i) Vt[(vc + i) * VSTR + vr] = tv.e[i];
    }
#ifdef USE_ASYNC_LDS
    async_wait0();
#endif
    __syncthreads();

    // S = Q.K^T
    v8f s0 = {}, s1 = {};
    #pragma unroll
    for (int c = 0; c < 2; ++c) {
      frag_u u0, u1;
      const unsigned short* p0 = Ks + l16 * KSTR        + c * 32 + hi * 8;
      const unsigned short* p1 = Ks + (16 + l16) * KSTR + c * 32 + hi * 8;
      u0.q[0] = ((const uint4*)p0)[0]; u0.q[1] = ((const uint4*)(p0 + 16))[0];
      u1.q[0] = ((const uint4*)p1)[0]; u1.q[1] = ((const uint4*)(p1 + 16))[0];
      s0 = __builtin_amdgcn_wmma_f32_16x16x32_bf16(false, aq[c], false,
              __builtin_bit_cast(v16bf, u0.v), (short)0, s0, false, false);
      s1 = __builtin_amdgcn_wmma_f32_16x16x32_bf16(false, aq[c], false,
              __builtin_bit_cast(v16bf, u1.v), (short)0, s1, false, false);
    }

    // scale + gate bias + causal mask + online softmax
    #pragma unroll
    for (int r = 0; r < 8; ++r) {
      const int q = qbase + r + hi * 8;
      float v0 = s0[r] * 0.125f + lg[r];
      float v1 = s1[r] * 0.125f + lg[r];
      if (k0 + l16 > q)      v0 = -3.0e38f;
      if (k0 + 16 + l16 > q) v1 = -3.0e38f;
      float rm = fmaxf(v0, v1);
      #pragma unroll
      for (int off = 8; off >= 1; off >>= 1)
        rm = fmaxf(rm, __shfl_xor(rm, off, 16));
      const float mnew  = fmaxf(Mr[r], rm);
      const float alpha = __expf(Mr[r] - mnew);
      const float p0 = __expf(v0 - mnew);
      const float p1 = __expf(v1 - mnew);
      float rs = p0 + p1;
      #pragma unroll
      for (int off = 8; off >= 1; off >>= 1)
        rs += __shfl_xor(rs, off, 16);
      Lr[r] = Lr[r] * alpha + rs;
      Mr[r] = mnew;
      #pragma unroll
      for (int dt = 0; dt < 4; ++dt) O[dt][r] *= alpha;
      Pt[wave][r + hi * 8][l16]      = p0;   // C-layout -> LDS transpose
      Pt[wave][r + hi * 8][16 + l16] = p1;
    }
    __syncthreads();

    // P as A-fragment (16x32 bf16)
    v16u pu;
    #pragma unroll
    for (int r = 0; r < 8; ++r) {
      const int kk = 2 * (r & 3) + ((r >= 4) ? 16 : 0) + hi * 8;
      pu[2 * r]     = f2bf(Pt[wave][l16][kk]);
      pu[2 * r + 1] = f2bf(Pt[wave][l16][kk + 1]);
    }
    const v16bf pa = __builtin_bit_cast(v16bf, pu);

    // O(16x64) += P(16x32) . V(32x64)
    #pragma unroll
    for (int dt = 0; dt < 4; ++dt) {
      frag_u u;
      const unsigned short* p = Vt + (dt * 16 + l16) * VSTR + hi * 8;
      u.q[0] = ((const uint4*)p)[0]; u.q[1] = ((const uint4*)(p + 16))[0];
      O[dt] = __builtin_amdgcn_wmma_f32_16x16x32_bf16(false, pa, false,
                 __builtin_bit_cast(v16bf, u.v), (short)0, O[dt], false, false);
    }
    __syncthreads();
  }

  #pragma unroll
  for (int dt = 0; dt < 4; ++dt)
    #pragma unroll
    for (int r = 0; r < 8; ++r) {
      const int q = qbase + r + hi * 8;
      y[((size_t)b * T_ + q) * D_ + h * HD_ + dt * 16 + l16] =
          f2bf(O[dt][r] / Lr[r]);
    }
}

// =====================================================================
// Router: thread per token -> 9 logits, softmax, top-2
// =====================================================================
__global__ __launch_bounds__(256) void k_router(
    const float* __restrict__ h2f, const float* __restrict__ rw,
    int* __restrict__ topi, float* __restrict__ topp, float* __restrict__ wid)
{
  const int t = blockIdx.x * 256 + threadIdx.x;
  if (t >= BT_) return;
  float lg[E_ + 1];
  #pragma unroll
  for (int j = 0; j <= E_; ++j) lg[j] = 0.f;
  const float* hr = h2f + (size_t)t * D_;
  for (int d = 0; d < D_; ++d) {
    const float hv = hr[d];
    const float* r = rw + d * (E_ + 1);
    #pragma unroll
    for (int j = 0; j <= E_; ++j) lg[j] += hv * r[j];
  }
  float mx = lg[0];
  for (int j = 1; j <= E_; ++j) mx = fmaxf(mx, lg[j]);
  float p[E_ + 1], s = 0.f;
  for (int j = 0; j <= E_; ++j) { p[j] = __expf(lg[j] - mx); s += p[j]; }
  const float inv = 1.f / s;
  for (int j = 0; j <= E_; ++j) p[j] *= inv;
  int i0 = 0;
  for (int j = 1; j <= E_; ++j) if (p[j] > p[i0]) i0 = j;
  int i1 = (i0 == 0) ? 1 : 0;
  for (int j = 0; j <= E_; ++j) if (j != i0 && p[j] > p[i1]) i1 = j;
  topi[t * 2]     = i0; topp[t * 2]     = p[i0];
  topi[t * 2 + 1] = i1; topp[t * 2 + 1] = p[i1];
  float w = 0.f;
  if (i0 == E_) w += p[i0];
  if (i1 == E_) w += p[i1];
  wid[t] = w;
}

// out = x1 + w_identity * h2  (residual + dummy identity expert)
__global__ __launch_bounds__(256) void k_init_out(
    const float* __restrict__ x1, const float* __restrict__ h2f,
    const float* __restrict__ wid, float* __restrict__ out, int n)
{
  const int i = blockIdx.x * 256 + threadIdx.x;
  if (i >= n) return;
  out[i] = x1[i] + wid[i / D_] * h2f[i];
}

// Build per-expert token lists
__global__ __launch_bounds__(256) void k_listbuild(
    const int* __restrict__ topi, const float* __restrict__ topp,
    int* __restrict__ cnt, int* __restrict__ list, float* __restrict__ lw)
{
  const int t = blockIdx.x * 256 + threadIdx.x;
  if (t >= BT_) return;
  #pragma unroll
  for (int s = 0; s < 2; ++s) {
    const int   j = topi[t * 2 + s];
    const float p = topp[t * 2 + s];
    if (j < E_) {
      const int pos = atomicAdd(&cnt[j], 1);
      list[(size_t)j * BT_ + pos] = t;
      lw[(size_t)j * BT_ + pos]   = p;
    }
  }
}

// =====================================================================
extern "C" void kernel_launch(void* const* d_in, const int* in_sizes, int n_in,
                              void* d_out, int out_size, void* d_ws, size_t ws_size,
                              hipStream_t stream)
{
  (void)in_sizes; (void)n_in; (void)out_size; (void)ws_size;
  const float* x     = (const float*)d_in[0];
  const float* gate  = (const float*)d_in[1];
  const float* ln1w  = (const float*)d_in[2];
  const float* ln2w  = (const float*)d_in[3];
  const float* Wqkv  = (const float*)d_in[4];
  const float* Wproj = (const float*)d_in[5];
  const float* rW    = (const float*)d_in[6];
  const float* Wfc   = (const float*)d_in[7];
  const float* Wpj   = (const float*)d_in[8];
  float* out = (float*)d_out;
  char* ws = (char*)d_ws;

  size_t off = 0;
  auto alloc = [&](size_t bytes) {
    size_t o = off; off = (off + bytes + 255) & ~(size_t)255; return o;
  };
  unsigned short* wqkvT  = (unsigned short*)(ws + alloc((size_t)3 * D_ * D_ * 2));
  unsigned short* wprojT = (unsigned short*)(ws + alloc((size_t)D_ * D_ * 2));
  unsigned short* wfcT   = (unsigned short*)(ws + alloc((size_t)E_ * D_ * 4 * D_ * 2));
  unsigned short* wpjT   = (unsigned short*)(ws + alloc((size_t)E_ * 4 * D_ * D_ * 2));
  unsigned short* h1_bf  = (unsigned short*)(ws + alloc((size_t)BT_ * D_ * 2));
  unsigned short* qkv_bf = (unsigned short*)(ws + alloc((size_t)BT_ * 3 * D_ * 2));
  unsigned short* y_bf   = (unsigned short*)(ws + alloc((size_t)BT_ * D_ * 2));
  float*          x1     = (float*)(ws + alloc((size_t)BT_ * D_ * 4));
  float*          h2f    = (float*)(ws + alloc((size_t)BT_ * D_ * 4));
  unsigned short* h2_bf  = (unsigned short*)(ws + alloc((size_t)BT_ * D_ * 2));
  int*            topi   = (int*)(ws + alloc((size_t)BT_ * 2 * 4));
  float*          topp   = (float*)(ws + alloc((size_t)BT_ * 2 * 4));
  float*          wid    = (float*)(ws + alloc((size_t)BT_ * 4));
  int*            cnt    = (int*)(ws + alloc(256));
  int*            list   = (int*)(ws + alloc((size_t)E_ * BT_ * 4));
  float*          lw     = (float*)(ws + alloc((size_t)E_ * BT_ * 4));
  unsigned short* hfc_bf = (unsigned short*)(ws + alloc((size_t)BT_ * 4 * D_ * 2));

  // transposed bf16 weights: Bt[n][k]
  k_cvt_t<<<dim3(D_ / 32, 3 * D_ / 32), 256, 0, stream>>>(Wqkv, wqkvT, D_, 3 * D_);
  k_cvt_t<<<dim3(D_ / 32, D_ / 32), 256, 0, stream>>>(Wproj, wprojT, D_, D_);
  k_cvt_t<<<dim3(D_ / 32, 4 * D_ / 32, E_), 256, 0, stream>>>(Wfc, wfcT, D_, 4 * D_);
  k_cvt_t<<<dim3(4 * D_ / 32, D_ / 32, E_), 256, 0, stream>>>(Wpj, wpjT, 4 * D_, D_);
  (void)hipMemsetAsync(cnt, 0, 256, stream);

  // attention sublayer
  k_ln<<<BT_, 256, 0, stream>>>(x, ln1w, h1_bf, nullptr);
  k_gemm_bf16<<<dim3(BT_ / BM, 3 * D_ / BN), 128, 0, stream>>>(
      h1_bf, D_, wqkvT, D_, nullptr, qkv_bf, 3 * D_,
      BT_, 3 * D_, D_, nullptr, nullptr, nullptr, nullptr, nullptr, 0);
  k_attn<<<B_ * H_ * (T_ / 64), 128, 0, stream>>>(qkv_bf, gate, y_bf);
  k_gemm_bf16<<<dim3(BT_ / BM, D_ / BN), 128, 0, stream>>>(
      y_bf, D_, wprojT, D_, x1, nullptr, D_,
      BT_, D_, D_, nullptr, nullptr, nullptr, nullptr, x, 0);

  // MoE sublayer
  k_ln<<<BT_, 256, 0, stream>>>(x1, ln2w, h2_bf, h2f);
  k_router<<<BT_ / 256, 256, 0, stream>>>(h2f, rW, topi, topp, wid);
  k_init_out<<<(BT_ * D_) / 256, 256, 0, stream>>>(x1, h2f, wid, out, BT_ * D_);
  k_listbuild<<<BT_ / 256, 256, 0, stream>>>(topi, topp, cnt, list, lw);

  for (int j = 0; j < E_; ++j) {
    // hfc = gelu(gather(h2) @ W_fc[j])  (compact rows, bf16 out)
    k_gemm_bf16<<<dim3(BT_ / BM, 4 * D_ / BN), 128, 0, stream>>>(
        h2_bf, D_, wfcT + (size_t)j * D_ * 4 * D_, D_,
        nullptr, hfc_bf, 4 * D_, BT_, 4 * D_, D_,
        list + (size_t)j * BT_, nullptr, nullptr, cnt + j, nullptr, 1);
    // out[token] += p * (hfc @ W_pj[j])  (scatter accumulate)
    k_gemm_bf16<<<dim3(BT_ / BM, D_ / BN), 128, 0, stream>>>(
        hfc_bf, 4 * D_, wpjT + (size_t)j * 4 * D_ * D_, 4 * D_,
        out, nullptr, D_, BT_, D_, 4 * D_,
        nullptr, list + (size_t)j * BT_, lw + (size_t)j * BT_, cnt + j, nullptr, 0);
  }
}
